// SatCamDeformAttnFusion_76957224009862
// MI455X (gfx1250) — matrix operand order repaired and checked by reading, not compiled, not run on hardware
//
#include <hip/hip_runtime.h>
#include <cstddef>
#include <cstdint>

// ---------------------------------------------------------------------------
// MI455X / gfx1250: wave32, WMMA bf16 16x16x32 (f32 accum),
// double-buffered async global->LDS copies overlapped with compute.
// ---------------------------------------------------------------------------
typedef __bf16 bf16;
typedef __attribute__((ext_vector_type(16))) __bf16 bf16x16;
typedef __attribute__((ext_vector_type(8)))  float  f32x8;

#define BB   2
#define HH   128
#define WW   128
#define NN   16384            // H*W
#define CCH  256
#define MR   32768            // B*N rows
#define NHH  8
#define HD   32

// LDS byte offset of a __shared__ object (flat LDS aperture: low 32 bits).
__device__ __forceinline__ unsigned lds_off(const void* p) {
  return (unsigned)(unsigned long long)(uintptr_t)p;
}

// gfx1250 async copy: 16 bytes global -> LDS per lane (ASYNCcnt tracked).
__device__ __forceinline__ void async_b128(unsigned ldsb, const void* g) {
  asm volatile("global_load_async_to_lds_b128 %0, %1, off"
               :: "v"(ldsb), "v"((unsigned long long)(uintptr_t)g) : "memory");
}
__device__ __forceinline__ void wait_async0() {
  asm volatile("s_wait_asynccnt 0x0" ::: "memory");
}

// ===========================================================================
// Generic bf16 WMMA GEMM:  out(M,N) = act(A(M,K) @ W + bias [+ resid])
//   A  : (M,K) row-major bf16,  Wn : N-MAJOR weights (N rows x K cols) bf16
// Block tile 64(M) x 64(N), 8 waves as 4(M) x 2(N); each wave: 16x32 output
// = 2 WMMAs sharing one A fragment.  K-step 32, LDS double-buffered, async
// copies for tile i+1 issued before computing tile i (latency hidden).
// act: 0 none, 1 relu, 2 sigmoid.  outF/outB/bias/resid nullable.
// ===========================================================================
__global__ __launch_bounds__(256)
void k_gemm_bf16(const bf16* __restrict__ A, const bf16* __restrict__ Wn,
                 const float* __restrict__ bias, const float* __restrict__ resid,
                 float* __restrict__ outF, bf16* __restrict__ outB,
                 int Ktot, int Ntot, int act)
{
  __shared__ bf16 As[2][64][40];    // [buf][M row][32 K + pad]
  __shared__ bf16 BsT[2][64][40];   // [buf][N row][32 K + pad]

  const int tid  = threadIdx.x;
  const int m0   = blockIdx.x * 64;
  const int n0   = blockIdx.y * 64;
  const int lane = tid & 31;
  const int wid  = tid >> 5;
  const int wm   = wid & 3;         // 4 M-subtiles of 16
  const int wn   = wid >> 2;        // 2 N-subtiles of 32
  const int half = lane >> 4;
  const int ml   = lane & 15;
  const int kb   = half * 8;        // ISA 16-bit A-fragment K base

  // staging: each of 256 threads copies one 16B chunk of A and one of B
  const int sr  = tid >> 2;                       // row 0..63
  const int sc  = (tid & 3) << 3;                 // bf16 col in 32-K tile
  const unsigned stg = (unsigned)(sr * 80 + (tid & 3) * 16);
  const unsigned ldsA[2] = { lds_off(&As[0][0][0])  + stg, lds_off(&As[1][0][0])  + stg };
  const unsigned ldsB[2] = { lds_off(&BsT[0][0][0]) + stg, lds_off(&BsT[1][0][0]) + stg };
  int brow = n0 + sr; if (brow >= Ntot) brow = Ntot - 1;   // clamp, stores guarded
  const bf16* agp = A  + (size_t)(m0 + sr) * Ktot + sc;
  const bf16* bgp = Wn + (size_t)brow * Ktot + sc;

  f32x8 acc0 = {}, acc1 = {};
  const int nk = Ktot >> 5;

  // prologue: tile 0 in flight
  async_b128(ldsA[0], agp);
  async_b128(ldsB[0], bgp);

  for (int i = 0; i < nk; ++i) {
    const int cur = i & 1, nxt = cur ^ 1;
    wait_async0();                    // this wave's copies for tile i landed
    __syncthreads();                  // all waves' copies landed; prev readers done
    if (i + 1 < nk) {                 // overlap: fetch tile i+1 during compute
      async_b128(ldsA[nxt], agp + (i + 1) * 32);
      async_b128(ldsB[nxt], bgp + (i + 1) * 32);
      if (lane == 0 && (i + 2) < nk)
        __builtin_prefetch(agp + (i + 2) * 32, 0, 1);   // global_prefetch_b8
    }

    bf16x16 af, b0, b1;
    const bf16* ra  = &As[cur][wm * 16 + ml][0];
    const bf16* rb0 = &BsT[cur][wn * 32 + ml][0];
    const bf16* rb1 = &BsT[cur][wn * 32 + 16 + ml][0];
#pragma unroll
    for (int j = 0; j < 16; ++j) {    // A: contiguous 32B per lane
      af[j] = ra[half * 16 + j];
    }
#pragma unroll
    for (int j = 0; j < 16; ++j) {    // B: contiguous 32B per lane
      b0[j] = rb0[half * 16 + j];
      b1[j] = rb1[half * 16 + j];
    }
    acc0 = __builtin_amdgcn_wmma_f32_16x16x32_bf16(false, af, false, b0,
                                                   (short)0, acc0, false, false);
    acc1 = __builtin_amdgcn_wmma_f32_16x16x32_bf16(false, af, false, b1,
                                                   (short)0, acc1, false, false);
  }

  const int ng0 = n0 + wn * 32 + ml;
  const int ng1 = ng0 + 16;
  const float bc0 = (bias && ng0 < Ntot) ? bias[ng0] : 0.0f;
  const float bc1 = (bias && ng1 < Ntot) ? bias[ng1] : 0.0f;
#pragma unroll
  for (int r = 0; r < 8; ++r) {
    const int mg = m0 + wm * 16 + half * 8 + r;
    float v0 = acc0[r] + bc0;
    float v1 = acc1[r] + bc1;
    if (resid) {
      if (ng0 < Ntot) v0 += resid[(size_t)mg * Ntot + ng0];
      if (ng1 < Ntot) v1 += resid[(size_t)mg * Ntot + ng1];
    }
    if (act == 1)      { v0 = fmaxf(v0, 0.0f); v1 = fmaxf(v1, 0.0f); }
    else if (act == 2) { v0 = 1.0f / (1.0f + __expf(-v0));
                         v1 = 1.0f / (1.0f + __expf(-v1)); }
    if (outF) {
      if (ng0 < Ntot) outF[(size_t)mg * Ntot + ng0] = v0;
      if (ng1 < Ntot) outF[(size_t)mg * Ntot + ng1] = v1;
    }
    if (outB) {
      if (ng0 < Ntot) outB[(size_t)mg * Ntot + ng0] = (bf16)v0;
      if (ng1 < Ntot) outB[(size_t)mg * Ntot + ng1] = (bf16)v1;
    }
  }
}

// zero-padded im2col 16B chunk load for the conv A tile
__device__ __forceinline__ uint4 conv_loadA(const bf16* __restrict__ gi,
                                            int pb, int pyy, int pxx,
                                            int sc, int k0)
{
  const int kk = k0 >> 9;               // ky*3+kx
  const int ky = kk / 3;
  const int kx = kk - ky * 3;
  const int c0 = k0 & 511;
  const int y = pyy + ky - 1;
  const int x = pxx + kx - 1;
  uint4 d = {0u, 0u, 0u, 0u};
  if ((unsigned)y < (unsigned)HH && (unsigned)x < (unsigned)WW)
    d = *(const uint4*)&gi[(((size_t)(pb * HH + y) * WW + x) * 512) + c0 + sc];
  return d;
}

// ===========================================================================
// Implicit-GEMM 3x3 conv (SAME, pad 1): gi NHWC bf16 (B,H,W,512),
// weights N-major (128 rows x 4608 cols), k=(ky*3+kx)*512+c. BN+ReLU fused.
// Tile 64 pixels x 64 out-channels; B async double-buffered, A register-
// pipelined (next tile loaded during compute, stored to LDS next iter).
// ===========================================================================
__global__ __launch_bounds__(256)
void k_conv3x3_bf16(const bf16* __restrict__ gi, const bf16* __restrict__ W2n,
                    const float* __restrict__ bng, const float* __restrict__ bnb,
                    bf16* __restrict__ hout)
{
  __shared__ bf16 As[64][40];
  __shared__ bf16 BsT[2][64][40];

  const int tid  = threadIdx.x;
  const int m0   = blockIdx.x * 64;
  const int n0   = blockIdx.y * 64;
  const int lane = tid & 31;
  const int wid  = tid >> 5;
  const int wm   = wid & 3;
  const int wn   = wid >> 2;
  const int half = lane >> 4;
  const int ml   = lane & 15;
  const int kb   = half * 8;

  const int sr = tid >> 2;
  const int sc = (tid & 3) << 3;
  const unsigned stg = (unsigned)(sr * 80 + (tid & 3) * 16);
  const unsigned ldsB[2] = { lds_off(&BsT[0][0][0]) + stg, lds_off(&BsT[1][0][0]) + stg };
  const bf16* bgp = W2n + (size_t)(n0 + sr) * 4608 + sc;

  const int pixel = m0 + sr;            // A staging pixel for this thread
  const int pb  = pixel >> 14;
  const int rem = pixel & (NN - 1);
  const int pyy = rem >> 7;
  const int pxx = rem & 127;

  f32x8 acc0 = {}, acc1 = {};
  const int nk = 4608 >> 5;             // 144 K-steps

  // prologue: B tile 0 async, A tile 0 in registers
  async_b128(ldsB[0], bgp);
  uint4 areg = conv_loadA(gi, pb, pyy, pxx, sc, 0);

  for (int i = 0; i < nk; ++i) {
    const int cur = i & 1, nxt = cur ^ 1;
    wait_async0();
    __syncthreads();                    // prev readers of As / BsT[nxt] done
    *(uint4*)((bf16*)&As[0][0] + sr * 40 + sc) = areg;   // A tile i -> LDS
    if (i + 1 < nk) {
      async_b128(ldsB[nxt], bgp + (i + 1) * 32);
      areg = conv_loadA(gi, pb, pyy, pxx, sc, (i + 1) * 32);  // overlap
    }
    __syncthreads();                    // A tile visible

    bf16x16 af, b0, b1;
    const bf16* ra  = &As[wm * 16 + ml][0];
    const bf16* rb0 = &BsT[cur][wn * 32 + ml][0];
    const bf16* rb1 = &BsT[cur][wn * 32 + 16 + ml][0];
#pragma unroll
    for (int j = 0; j < 16; ++j) { af[j] = ra[half * 16 + j]; }
#pragma unroll
    for (int j = 0; j < 16; ++j) { b0[j] = rb0[half * 16 + j]; b1[j] = rb1[half * 16 + j]; }

    acc0 = __builtin_amdgcn_wmma_f32_16x16x32_bf16(false, af, false, b0,
                                                   (short)0, acc0, false, false);
    acc1 = __builtin_amdgcn_wmma_f32_16x16x32_bf16(false, af, false, b1,
                                                   (short)0, acc1, false, false);
  }

  const int ng0 = n0 + wn * 32 + ml;    // output channels (always < 128)
  const int ng1 = ng0 + 16;
  const float scl = 0.9999950000374996f;  // rsqrt(1 + 1e-5)
  const float g0 = bng[ng0] * scl, b0v = bnb[ng0];
  const float g1 = bng[ng1] * scl, b1v = bnb[ng1];
#pragma unroll
  for (int r = 0; r < 8; ++r) {
    const int mg = m0 + wm * 16 + half * 8 + r;
    hout[(size_t)mg * 128 + ng0] = (bf16)fmaxf(acc0[r] * g0 + b0v, 0.0f);
    hout[(size_t)mg * 128 + ng1] = (bf16)fmaxf(acc1[r] * g1 + b1v, 0.0f);
  }
}

// ===========================================================================
// Deformable bilinear sampling. One wave per (b, n, head); lane = channel.
// ===========================================================================
__global__ __launch_bounds__(256)
void k_deform_sample(const float* __restrict__ val, const float* __restrict__ off,
                     const float* __restrict__ awl, bf16* __restrict__ samp)
{
  const int gw   = blockIdx.x * 8 + (threadIdx.x >> 5);
  const int lane = threadIdx.x & 31;
  const int h = gw & 7;
  const int n = (gw >> 3) & (NN - 1);
  const int b = gw >> 17;
  const int py = n >> 7, px = n & 127;
  const size_t row = (size_t)b * NN + n;

  const float* op = off + row * 64 + h * 8;
  const float* lp = awl + row * 32 + h * 4;
  float l0 = lp[0], l1 = lp[1], l2 = lp[2], l3 = lp[3];
  float mx = fmaxf(fmaxf(l0, l1), fmaxf(l2, l3));
  float e0 = __expf(l0 - mx), e1 = __expf(l1 - mx),
        e2 = __expf(l2 - mx), e3 = __expf(l3 - mx);
  float inv = 1.0f / (e0 + e1 + e2 + e3);
  float ew[4] = { e0 * inv, e1 * inv, e2 * inv, e3 * inv };

  const float* vb = val + (size_t)b * NN * CCH + h * HD + lane;
  float accv = 0.0f;
#pragma unroll
  for (int p = 0; p < 4; ++p) {
    float fx = (float)px + op[2 * p];          // ((px+.5)/W + off/W)*W - .5
    float fy = (float)py + op[2 * p + 1];
    float x0f = floorf(fx), y0f = floorf(fy);
    float wx = fx - x0f, wy = fy - y0f;
    int x0 = (int)x0f, y0 = (int)y0f;
    float aw  = ew[p];
    float w00 = aw * (1.0f - wx) * (1.0f - wy);
    float w10 = aw * wx * (1.0f - wy);
    float w01 = aw * (1.0f - wx) * wy;
    float w11 = aw * wx * wy;
    if ((unsigned)x0     < 128u && (unsigned)y0     < 128u) accv += w00 * vb[(size_t)(y0 * WW + x0) * CCH];
    if ((unsigned)(x0+1) < 128u && (unsigned)y0     < 128u) accv += w10 * vb[(size_t)(y0 * WW + x0 + 1) * CCH];
    if ((unsigned)x0     < 128u && (unsigned)(y0+1) < 128u) accv += w01 * vb[(size_t)((y0+1) * WW + x0) * CCH];
    if ((unsigned)(x0+1) < 128u && (unsigned)(y0+1) < 128u) accv += w11 * vb[(size_t)((y0+1) * WW + x0 + 1) * CCH];
  }
  samp[row * CCH + h * HD + lane] = (bf16)accv;
}

// ===========================================================================
// LayerNorm over C=256, one wave32 per row; optional NHWC bf16 scatter.
// ===========================================================================
__global__ __launch_bounds__(256)
void k_ln(const float* __restrict__ x, const float* __restrict__ g,
          const float* __restrict__ b, float* __restrict__ outF,
          bf16* __restrict__ gi, int choff)
{
  const int row  = blockIdx.x * 8 + (threadIdx.x >> 5);
  const int lane = threadIdx.x & 31;
  const int cb   = lane * 8;
  const float* xr = x + (size_t)row * CCH;
  float v[8], s = 0.0f, ss = 0.0f;
#pragma unroll
  for (int j = 0; j < 8; ++j) { v[j] = xr[cb + j]; s += v[j]; ss += v[j] * v[j]; }
#pragma unroll
  for (int o = 16; o > 0; o >>= 1) { s += __shfl_xor(s, o, 32); ss += __shfl_xor(ss, o, 32); }
  float m   = s * (1.0f / 256.0f);
  float var = ss * (1.0f / 256.0f) - m * m;
  float inv = rsqrtf(var + 1e-5f);
#pragma unroll
  for (int j = 0; j < 8; ++j) {
    float y = (v[j] - m) * inv * g[cb + j] + b[cb + j];
    if (outF) outF[(size_t)row * CCH + cb + j] = y;
    if (gi)   gi[(size_t)row * 512 + choff + cb + j] = (bf16)y;
  }
}

// Final LN: NCHW output + cam_bev residual.
__global__ __launch_bounds__(256)
void k_ln_final(const float* __restrict__ x, const float* __restrict__ g,
                const float* __restrict__ b, const float* __restrict__ cam,
                float* __restrict__ out)
{
  const int row  = blockIdx.x * 8 + (threadIdx.x >> 5);
  const int lane = threadIdx.x & 31;
  const int cb   = lane * 8;
  const int bb   = row >> 14;
  const int n    = row & (NN - 1);
  const float* xr = x + (size_t)row * CCH;
  float v[8], s = 0.0f, ss = 0.0f;
#pragma unroll
  for (int j = 0; j < 8; ++j) { v[j] = xr[cb + j]; s += v[j]; ss += v[j] * v[j]; }
#pragma unroll
  for (int o = 16; o > 0; o >>= 1) { s += __shfl_xor(s, o, 32); ss += __shfl_xor(ss, o, 32); }
  float m   = s * (1.0f / 256.0f);
  float var = ss * (1.0f / 256.0f) - m * m;
  float inv = rsqrtf(var + 1e-5f);
#pragma unroll
  for (int j = 0; j < 8; ++j) {
    float y = (v[j] - m) * inv * g[cb + j] + b[cb + j];
    size_t idx = ((size_t)(bb * CCH + cb + j)) * NN + n;
    out[idx] = y + cam[idx];
  }
}

// ---------------------------- layout / cast helpers ------------------------
__global__ void k_to_nc(const float* __restrict__ src, float* __restrict__ dF,
                        bf16* __restrict__ dB)
{
  size_t i = (size_t)blockIdx.x * 256 + threadIdx.x;
  if (i >= (size_t)MR * CCH) return;
  int c = (int)(i & 255);
  int n = (int)((i >> 8) & (NN - 1));
  int b = (int)(i >> 22);
  float v = src[((size_t)(b * CCH + c)) * NN + n];
  dF[i] = v; dB[i] = (bf16)v;
}

__global__ void k_wcast(const float* __restrict__ s, bf16* __restrict__ d, int n)
{
  int i = blockIdx.x * 256 + threadIdx.x;
  if (i < n) d[i] = (bf16)s[i];
}

// (K,N) row-major f32 -> N-major (N rows x K cols) bf16
__global__ void k_wT(const float* __restrict__ s, bf16* __restrict__ d, int K, int N)
{
  int i = blockIdx.x * 256 + threadIdx.x;
  if (i >= K * N) return;
  int k = i / N, n = i - k * N;
  d[(size_t)n * K + k] = (bf16)s[i];
}

// g1_w OIHW (128,512,3,3) -> N-major (o rows, k=(ky*3+kx)*512+c cols) bf16
__global__ void k_convw(const float* __restrict__ s, bf16* __restrict__ d)
{
  int i = blockIdx.x * 256 + threadIdx.x;
  if (i >= 128 * 4608) return;
  int o = i / 4608, rem = i - o * 4608;
  int c = rem / 9, r9 = rem - c * 9;
  d[(size_t)o * 4608 + r9 * 512 + c] = (bf16)s[i];
}

__global__ void k_fuse(const float* __restrict__ g, const float* __restrict__ cm,
                       const float* __restrict__ sm, float* __restrict__ fF,
                       bf16* __restrict__ fB)
{
  size_t i = (size_t)blockIdx.x * 256 + threadIdx.x;
  if (i >= (size_t)MR * CCH) return;
  float gg = g[i];
  float v = gg * cm[i] + (1.0f - gg) * sm[i];
  fF[i] = v; fB[i] = (bf16)v;
}

// ===========================================================================
extern "C" void kernel_launch(void* const* d_in, const int* in_sizes, int n_in,
                              void* d_out, int out_size, void* d_ws, size_t ws_size,
                              hipStream_t stream)
{
  (void)in_sizes; (void)n_in; (void)out_size; (void)ws_size;
  const float* cam_bev = (const float*)d_in[0];
  const float* sat_bev = (const float*)d_in[1];
  const float* c_off_w = (const float*)d_in[2];  const float* c_off_b = (const float*)d_in[3];
  const float* c_aw_w  = (const float*)d_in[4];  const float* c_aw_b  = (const float*)d_in[5];
  const float* c_vp_w  = (const float*)d_in[6];  const float* c_vp_b  = (const float*)d_in[7];
  const float* c_op_w  = (const float*)d_in[8];  const float* c_op_b  = (const float*)d_in[9];
  const float* s_off_w = (const float*)d_in[10]; const float* s_off_b = (const float*)d_in[11];
  const float* s_aw_w  = (const float*)d_in[12]; const float* s_aw_b  = (const float*)d_in[13];
  const float* s_vp_w  = (const float*)d_in[14]; const float* s_vp_b  = (const float*)d_in[15];
  const float* s_op_w  = (const float*)d_in[16]; const float* s_op_b  = (const float*)d_in[17];
  const float* ln1_g = (const float*)d_in[18]; const float* ln1_b = (const float*)d_in[19];
  const float* ln2_g = (const float*)d_in[20]; const float* ln2_b = (const float*)d_in[21];
  const float* lnf_g = (const float*)d_in[22]; const float* lnf_b = (const float*)d_in[23];
  const float* ffn1_w = (const float*)d_in[24]; const float* ffn1_b = (const float*)d_in[25];
  const float* ffn2_w = (const float*)d_in[26]; const float* ffn2_b = (const float*)d_in[27];
  const float* g1_w = (const float*)d_in[28];
  const float* bn_g = (const float*)d_in[29]; const float* bn_b = (const float*)d_in[30];
  const float* g2_w = (const float*)d_in[31]; const float* g2_b = (const float*)d_in[32];
  float* out = (float*)d_out;

  unsigned char* wsp = (unsigned char*)d_ws;
  size_t wsoff = 0;
  auto alloc = [&](size_t bytes) -> void* {
    void* p = wsp + wsoff;
    wsoff += (bytes + 255) & ~(size_t)255;
    return p;
  };
  const size_t F  = (size_t)MR * CCH * sizeof(float);
  const size_t Bh = (size_t)MR * CCH * sizeof(bf16);

  float* qcam_f  = (float*)alloc(F);
  float* qsat_f  = (float*)alloc(F);
  bf16*  qcam_b  = (bf16*)alloc(Bh);
  bf16*  qsat_b  = (bf16*)alloc(Bh);
  float* camatt  = (float*)alloc(F);
  float* satatt  = (float*)alloc(F);
  bf16*  gi      = (bf16*)alloc((size_t)MR * 512 * sizeof(bf16));
  float* val_f   = (float*)alloc(F);
  float* off_f   = (float*)alloc((size_t)MR * 64 * sizeof(float));
  float* awl_f   = (float*)alloc((size_t)MR * 32 * sizeof(float));
  bf16*  samp_b  = (bf16*)alloc(Bh);
  float* att_f   = (float*)alloc(F);
  bf16*  h_b     = (bf16*)alloc((size_t)MR * 128 * sizeof(bf16));
  float* g_f     = (float*)alloc(F);
  float* fused_f = (float*)alloc(F);
  bf16*  fused_b = (bf16*)alloc(Bh);
  bf16*  t1_b    = (bf16*)alloc((size_t)MR * 512 * sizeof(bf16));
  float* ffn_f   = (float*)alloc(F);
  bf16* w_vp1  = (bf16*)alloc(256*256*2); bf16* w_off1 = (bf16*)alloc(256*64*2);
  bf16* w_aw1  = (bf16*)alloc(256*32*2);  bf16* w_op1  = (bf16*)alloc(256*256*2);
  bf16* w_vp2  = (bf16*)alloc(256*256*2); bf16* w_off2 = (bf16*)alloc(256*64*2);
  bf16* w_aw2  = (bf16*)alloc(256*32*2);  bf16* w_op2  = (bf16*)alloc(256*256*2);
  bf16* w_f1   = (bf16*)alloc(256*512*2); bf16* w_f2   = (bf16*)alloc(512*256*2);
  bf16* w_g2   = (bf16*)alloc(256*128*2); bf16* w_cv   = (bf16*)alloc(128*4608*2);

  // ---- weights -> N-major bf16 ----
  k_wT<<<(65536+255)/256,256,0,stream>>>(c_vp_w, w_vp1, 256, 256);
  k_wT<<<(16384+255)/256,256,0,stream>>>(c_off_w, w_off1, 256, 64);
  k_wT<<<(8192 +255)/256,256,0,stream>>>(c_aw_w, w_aw1, 256, 32);
  k_wT<<<(65536+255)/256,256,0,stream>>>(c_op_w, w_op1, 256, 256);
  k_wT<<<(65536+255)/256,256,0,stream>>>(s_vp_w, w_vp2, 256, 256);
  k_wT<<<(16384+255)/256,256,0,stream>>>(s_off_w, w_off2, 256, 64);
  k_wT<<<(8192 +255)/256,256,0,stream>>>(s_aw_w, w_aw2, 256, 32);
  k_wT<<<(65536+255)/256,256,0,stream>>>(s_op_w, w_op2, 256, 256);
  k_wT<<<(131072+255)/256,256,0,stream>>>(ffn1_w, w_f1, 256, 512);
  k_wT<<<(131072+255)/256,256,0,stream>>>(ffn2_w, w_f2, 512, 256);
  k_wcast<<<(32768+255)/256,256,0,stream>>>(g2_w, w_g2, 32768);   // (o,c) already N-major
  k_convw<<<(589824+255)/256,256,0,stream>>>(g1_w, w_cv);

  // ---- NCHW -> (M,C) f32 + bf16 ----
  k_to_nc<<<MR*CCH/256,256,0,stream>>>(cam_bev, qcam_f, qcam_b);
  k_to_nc<<<MR*CCH/256,256,0,stream>>>(sat_bev, qsat_f, qsat_b);

  dim3 blk(256);
  dim3 g256(MR/64, 4), g64(MR/64, 1), g512(MR/64, 8);

  // ---- c2s deformable attention (query=cam, kv=sat) ----
  k_gemm_bf16<<<g256,blk,0,stream>>>(qsat_b, w_vp1, c_vp_b, nullptr, val_f, nullptr, 256, 256, 0);
  k_gemm_bf16<<<g64 ,blk,0,stream>>>(qcam_b, w_off1, c_off_b, nullptr, off_f, nullptr, 256, 64, 0);
  k_gemm_bf16<<<g64 ,blk,0,stream>>>(qcam_b, w_aw1, c_aw_b, nullptr, awl_f, nullptr, 256, 32, 0);
  k_deform_sample<<<MR*NHH/8,blk,0,stream>>>(val_f, off_f, awl_f, samp_b);
  k_gemm_bf16<<<g256,blk,0,stream>>>(samp_b, w_op1, c_op_b, qcam_f, att_f, nullptr, 256, 256, 0);
  k_ln<<<MR/8,blk,0,stream>>>(att_f, ln1_g, ln1_b, camatt, gi, 0);

  // ---- s2c deformable attention (query=sat, kv=cam) ----
  k_gemm_bf16<<<g256,blk,0,stream>>>(qcam_b, w_vp2, s_vp_b, nullptr, val_f, nullptr, 256, 256, 0);
  k_gemm_bf16<<<g64 ,blk,0,stream>>>(qsat_b, w_off2, s_off_b, nullptr, off_f, nullptr, 256, 64, 0);
  k_gemm_bf16<<<g64 ,blk,0,stream>>>(qsat_b, w_aw2, s_aw_b, nullptr, awl_f, nullptr, 256, 32, 0);
  k_deform_sample<<<MR*NHH/8,blk,0,stream>>>(val_f, off_f, awl_f, samp_b);
  k_gemm_bf16<<<g256,blk,0,stream>>>(samp_b, w_op2, s_op_b, qsat_f, att_f, nullptr, 256, 256, 0);
  k_ln<<<MR/8,blk,0,stream>>>(att_f, ln2_g, ln2_b, satatt, gi, 256);

  // ---- 3x3 conv (implicit GEMM, fused BN+ReLU) ----
  k_conv3x3_bf16<<<dim3(MR/64,2),blk,0,stream>>>(gi, w_cv, bn_g, bn_b, h_b);

  // ---- gating projection + sigmoid ----
  k_gemm_bf16<<<g256,blk,0,stream>>>(h_b, w_g2, g2_b, nullptr, g_f, nullptr, 128, 256, 2);

  // ---- fuse ----
  k_fuse<<<MR*CCH/256,blk,0,stream>>>(g_f, camatt, satatt, fused_f, fused_b);

  // ---- FFN + residual ----
  k_gemm_bf16<<<g512,blk,0,stream>>>(fused_b, w_f1, ffn1_b, nullptr, nullptr, t1_b, 256, 512, 1);
  k_gemm_bf16<<<g256,blk,0,stream>>>(t1_b, w_f2, ffn2_b, fused_f, ffn_f, nullptr, 512, 256, 0);

  // ---- final LN + NCHW transpose + cam_bev residual ----
  k_ln_final<<<MR/8,blk,0,stream>>>(ffn_f, lnf_g, lnf_b, cam_bev, out);
}
